// _FlashRNNCudaFusedLayer_67860483277716
// MI455X (gfx1250) — compile-verified, matrix operand order checked by reading, not run on hardware
//
#include <hip/hip_runtime.h>

typedef __attribute__((ext_vector_type(16))) __bf16 v16bf;
typedef __attribute__((ext_vector_type(8)))  __bf16 v8bf;
typedef __attribute__((ext_vector_type(8)))  float  v8f;
typedef __attribute__((ext_vector_type(4)))  float  v4f;

#define SEQ_T 1024
#define NB    16
#define NHD   12
#define HD    64
#define NG    4

static __device__ __forceinline__ float fsigmoid(float x) {
    return 1.0f / (1.0f + __expf(-x));
}
static __device__ __forceinline__ float ftanh_fast(float x) {
    float e = __expf(2.0f * x);
    return (e - 1.0f) / (e + 1.0f);
}

// One workgroup per head (12 WGs), 4 waves (128 threads) per WG.
// wave w owns output columns o = w*16 .. w*16+15 for ALL 4 gates.
// R lives in registers as WMMA B-fragments for the entire sequence.
// c lives in the accumulator-fragment layout in registers.
// h bounces through double-buffered bf16 LDS (A-matrix source).
extern "C" __global__ __launch_bounds__(128, 1)
void flashlstm_wmma(const float* __restrict__ Wx,
                    const float* __restrict__ states0,
                    const float* __restrict__ R,
                    const float* __restrict__ bias,
                    float* __restrict__ out)
{
    __shared__ __align__(16) __bf16 h_lds[2][NB][HD];

    const int n    = blockIdx.x;           // head
    const int tid  = threadIdx.x;
    const int lane = tid & 31;
    const int wave = tid >> 5;             // 0..3 -> 16-column D tile

    const int colL  = lane & 15;
    const int oCol  = wave * 16 + colL;    // output column within head dim
    const int half8 = (lane >= 16) ? 8 : 0;
    const int rowA  = lane & 15;           // A-fragment row (batch)
    const int rowC0 = (lane >= 16) ? 8 : 0;// C-fragment row base (batch)

    const size_t planeOut = (size_t)(SEQ_T + 1) * NB * NHD * HD;

    // ---- Preload R as bf16 B-fragments: rf[gate][k-block], reused for 1024 steps ----
    // B-frag (32x16 bf16): lane (l&15) = column oCol; elems 0..7 <- K = kb*32+half8+{0..7},
    // elems 8..15 <- K = kb*32+16+half8+{0..7}. R rows are contiguous in d (K).
    v16bf rf[NG][2];
    #pragma unroll
    for (int g = 0; g < NG; ++g) {
        const float* Rrow = R + ((size_t)((n * NG + g) * HD + oCol)) * HD;
        #pragma unroll
        for (int kb = 0; kb < 2; ++kb) {
            const float* p0 = Rrow + kb * 32 + half8;
            const float* p1 = Rrow + kb * 32 + 16 + half8;
            v16bf f;
            #pragma unroll
            for (int e = 0; e < 8; ++e) {
                f[e]     = (__bf16)p0[e];
                f[e + 8] = (__bf16)p1[e];
            }
            rf[g][kb] = f;
        }
    }

    // Per-column bias, one float per gate (broadcast over batch rows)
    float bg[NG];
    #pragma unroll
    for (int g = 0; g < NG; ++g) bg[g] = bias[(n * NG + g) * HD + oCol];

    // ---- Init: h0 -> LDS buf 0 (bf16); emit t=0 states; c0 -> accumulator fragment ----
    for (int i = tid; i < NB * HD; i += 128) {
        int b = i >> 6, d = i & 63;
        float h0 = states0[(size_t)(0 * NB + b) * NHD * HD + n * HD + d];
        float c0 = states0[(size_t)(1 * NB + b) * NHD * HD + n * HD + d];
        h_lds[0][b][d] = (__bf16)h0;
        size_t base = (size_t)b * (NHD * HD) + n * HD + d;   // t = 0
        out[base]            = h0;
        out[planeOut + base] = c0;
    }

    v8f cfrag;
    #pragma unroll
    for (int r = 0; r < 8; ++r) {
        int b = rowC0 + r;
        cfrag[r] = states0[(size_t)(1 * NB + b) * NHD * HD + n * HD + oCol];
    }

    __syncthreads();

    for (int t = 0; t < SEQ_T; ++t) {
        const int buf = t & 1;

        // 1) Issue Wx loads early (consumed only after the WMMAs, so latency hides).
        //    G is innermost in memory -> one float4 per (b,oCol) gives all 4 gates.
        v4f wx[8];
        #pragma unroll
        for (int r = 0; r < 8; ++r) {
            int b = rowC0 + r;
            const float* p = Wx + (((size_t)t * NB + b) * NHD + n) * (HD * NG) + oCol * NG;
            wx[r] = __builtin_nontemporal_load((const v4f*)p);
        }

        // 2) A-fragments (h_t, bf16 16x32): two contiguous 16B runs per K-block.
        v16bf a0, a1;
        {
            const __bf16* hrow = &h_lds[buf][rowA][0];
            v8bf lo0 = *(const v8bf*)(hrow + 0  + half8);
            v8bf hi0 = *(const v8bf*)(hrow + 16 + half8);
            v8bf lo1 = *(const v8bf*)(hrow + 32 + half8);
            v8bf hi1 = *(const v8bf*)(hrow + 48 + half8);
            #pragma unroll
            for (int e = 0; e < 8; ++e) {
                a0[e] = lo0[e]; a0[e + 8] = hi0[e];
                a1[e] = lo1[e]; a1[e + 8] = hi1[e];
            }
        }

        // 3) Recurrent GEMM: 4 gates x (K=64 as two K=32 WMMAs)
        v8f acc[NG];
        #pragma unroll
        for (int g = 0; g < NG; ++g) {
            v8f z = {};
            z = __builtin_amdgcn_wmma_f32_16x16x32_bf16(false, a0, false, rf[g][0],
                                                        (short)0, z, false, false);
            z = __builtin_amdgcn_wmma_f32_16x16x32_bf16(false, a1, false, rf[g][1],
                                                        (short)0, z, false, false);
            acc[g] = z;
        }

        // 4) Gate activations + state update, fully element-wise in fragment layout
        v8f hfrag;
        #pragma unroll
        for (int r = 0; r < 8; ++r) {
            float gi = acc[0][r] + wx[r][0] + bg[0];
            float gf = acc[1][r] + wx[r][1] + bg[1];
            float gz = acc[2][r] + wx[r][2] + bg[2];
            float go = acc[3][r] + wx[r][3] + bg[3];
            float i_ = fsigmoid(gi);
            float f_ = fsigmoid(gf);
            float z_ = ftanh_fast(gz);
            float o_ = fsigmoid(go);
            float c_ = f_ * cfrag[r] + i_ * z_;
            cfrag[r] = c_;
            hfrag[r] = o_ * ftanh_fast(c_);
        }

        // 5) h -> other LDS buffer (bf16) for next step; h,c -> global (streaming NT)
        #pragma unroll
        for (int r = 0; r < 8; ++r) {
            int b = rowC0 + r;
            h_lds[buf ^ 1][b][oCol] = (__bf16)hfrag[r];
            size_t base = ((size_t)(t + 1) * NB + b) * (NHD * HD) + n * HD + oCol;
            __builtin_nontemporal_store(hfrag[r], out + base);
            __builtin_nontemporal_store(cfrag[r], out + planeOut + base);
        }
        __syncthreads();   // one barrier/step is enough with double-buffered h
    }
}

extern "C" void kernel_launch(void* const* d_in, const int* in_sizes, int n_in,
                              void* d_out, int out_size, void* d_ws, size_t ws_size,
                              hipStream_t stream) {
    (void)in_sizes; (void)n_in; (void)out_size; (void)d_ws; (void)ws_size;
    const float* Wx      = (const float*)d_in[0];
    const float* states0 = (const float*)d_in[1];
    const float* R       = (const float*)d_in[2];
    const float* b       = (const float*)d_in[3];
    float* out = (float*)d_out;
    flashlstm_wmma<<<dim3(NHD), dim3(128), 0, stream>>>(Wx, states0, R, b, out);
}
